// MultiHeadCasualSelfAttention_91104846283196
// MI455X (gfx1250) — matrix operand order repaired and checked by reading, not compiled, not run on hardware
//
#include <hip/hip_runtime.h>
#include <hip/hip_bf16.h>

// ---------------------------------------------------------------------------
// MultiHeadCausalSelfAttention for MI455X (gfx1250, wave32, WMMA bf16).
// B=4, S=2048, E=1024, N=16 heads, H=64.
// Pipeline: cvt/transpose -> QKV GEMM (WMMA + TDM-staged B tiles) ->
// flash attention (WMMA + online softmax, LDS C->A relayout) ->
// output GEMM (WMMA + TDM-staged B tiles).
// ---------------------------------------------------------------------------

#define BB 4
#define SS 2048
#define EE 1024
#define NH 16
#define HH 64
#define QKV3 3072      // 3*N*H
#define MROWS (BB*SS)  // 8192
#define KTILE 64       // K-slab per TDM stage
#define BROWPAD 72     // 64 elems + 8 pad (TDM pad: 4 DWORDs per 32 DWORDs)

typedef __bf16 bf16_t;
typedef __attribute__((ext_vector_type(16))) __bf16 v16bf;
typedef __attribute__((ext_vector_type(8)))  __bf16 v8bf;
typedef __attribute__((ext_vector_type(8)))  float  v8f;
typedef __attribute__((ext_vector_type(4)))  unsigned int v4u;
typedef __attribute__((ext_vector_type(8)))  int v8i;
typedef __attribute__((ext_vector_type(4)))  int v4i;

#if defined(__has_builtin)
#  if __has_builtin(__builtin_amdgcn_tensor_load_to_lds)
#    define USE_TDM 1
#  endif
#endif
#ifndef USE_TDM
#  define USE_TDM 0
#endif

// A-fragment (16x32 bf16, row-major source): this lane's 16-elem row chunk at p,
// elements 0..15 = p[0..15] (contiguous k order).
__device__ __forceinline__ v16bf load_frag_a(const bf16_t* p) {
  v8bf lo = *(const v8bf*)(p);
  v8bf hi = *(const v8bf*)(p + 8);
  return __builtin_shufflevector(lo, hi, 0,1,2,3,4,5,6,7,8,9,10,11,12,13,14,15);
}

// B-fragment: 16 contiguous bf16 per lane, 32B-aligned source (global).
__device__ __forceinline__ v16bf load_frag_b(const bf16_t* p) {
  return *(const v16bf*)p;
}

// B-fragment from 16B-aligned source (padded LDS rows): two 16B loads.
__device__ __forceinline__ v16bf load_frag_b16(const bf16_t* p) {
  v8bf lo = *(const v8bf*)(p);
  v8bf hi = *(const v8bf*)(p + 8);
  return __builtin_shufflevector(lo, hi, 0,1,2,3,4,5,6,7,8,9,10,11,12,13,14,15);
}

__device__ __forceinline__ v8f wmma_bf16(v16bf a, v16bf b, v8f c) {
  return __builtin_amdgcn_wmma_f32_16x16x32_bf16(
      /*neg_a=*/false, a, /*neg_b=*/false, b,
      /*c_mod=*/(short)0, c, /*reuse_a=*/false, /*reuse_b=*/false);
}

// --------------------- TDM: stage a 64x64 bf16 tile into LDS ----------------
// Source: 64 rows of 64 bf16, row stride = `stride` elements (row-major).
// Dest: LDS rows padded to 72 elems via TDM pad (interval 32 DW, amount 4 DW).
#if USE_TDM
__device__ __forceinline__ void tdm_load_tile(unsigned int lds_off,
                                              const bf16_t* gsrc,
                                              unsigned int stride_elems) {
  unsigned long long ga = (unsigned long long)(uintptr_t)gsrc;
  v4u g0;
  g0[0] = 1u;                                           // count=1, user mode
  g0[1] = lds_off;                                      // lds_addr (bytes)
  g0[2] = (unsigned int)(ga & 0xffffffffu);             // global_addr[31:0]
  g0[3] = (unsigned int)((ga >> 32) & 0x01ffffffu)      // global_addr[56:32]
          | (2u << 30);                                 // type=2 ("image")
  v8i g1;
  g1[0] = (int)0x07110000u;  // data_size=2B | pad_enable | interval=32DW | amount=4DW
  g1[1] = (int)(64u << 16);  // tensor_dim0 = 64  (bits 63:48)
  g1[2] = (int)(64u << 16);  // tensor_dim1 = 64  (bits 95:80)
  g1[3] = (int)(64u << 16);  // tile_dim0   = 64  (bits 127:112)
  g1[4] = 64;                // tile_dim1   = 64  (bits 143:128)
  g1[5] = (int)stride_elems; // tensor_dim0_stride[31:0] (bits 191:160)
  g1[6] = 0;
  g1[7] = 0;
  v4i z4 = {0, 0, 0, 0};
#if __clang_major__ >= 23
  v8i z8 = {0, 0, 0, 0, 0, 0, 0, 0};
  __builtin_amdgcn_tensor_load_to_lds(g0, g1, z4, z4, z8, 0);
#else
  __builtin_amdgcn_tensor_load_to_lds(g0, g1, z4, z4, 0);
#endif
}
#else
// Fallback: cooperative staging with vector loads (256 threads, 16 elems each).
__device__ __forceinline__ void stage_tile_sync(bf16_t* dst /*[64][BROWPAD]*/,
                                                const bf16_t* gsrc,
                                                unsigned int stride_elems,
                                                int tid) {
  const int r = tid >> 2;
  const int c = (tid & 3) * 16;
  v8bf lo = *(const v8bf*)(gsrc + (size_t)r * stride_elems + c);
  v8bf hi = *(const v8bf*)(gsrc + (size_t)r * stride_elems + c + 8);
  *(v8bf*)(dst + r * BROWPAD + c)     = lo;
  *(v8bf*)(dst + r * BROWPAD + c + 8) = hi;
}
#endif

// ------------------------- stage 0: conversions -----------------------------

__global__ void k_cvt_x(const float* __restrict__ x, bf16_t* __restrict__ xh, int n) {
  int i = blockIdx.x * blockDim.x + threadIdx.x;
  if (i < n) xh[i] = (bf16_t)x[i];
}

// W_qkv [E, 3NH] -> Wt [3NH, E]  (N-major so B-fragments load contiguously)
__global__ void k_trans_wqkv(const float* __restrict__ w, bf16_t* __restrict__ wt) {
  int i = blockIdx.x * blockDim.x + threadIdx.x;
  if (i < EE * QKV3) {
    int k = i / QKV3, c = i % QKV3;
    wt[(size_t)c * EE + k] = (bf16_t)w[i];
  }
}

// W_out [NH*H, E] -> Wt [E, NH*H]
__global__ void k_trans_wout(const float* __restrict__ w, bf16_t* __restrict__ wt) {
  int i = blockIdx.x * blockDim.x + threadIdx.x;
  if (i < EE * EE) {
    int k = i / EE, e = i % EE;
    wt[(size_t)e * EE + k] = (bf16_t)w[i];
  }
}

// ------------------------- stage 1: QKV projection --------------------------
// Workgroup = 8 waves covering a 128(M) x 64(N) macro-tile. The 64x64 bf16
// B-slab is staged into LDS once per block (TDM, double-buffered); each wave
// owns 16 M-rows. Q,K stored [B,N,S,H]; V stored transposed [B,N,H,S].

__global__ __launch_bounds__(256)
void k_qkv_gemm(const bf16_t* __restrict__ xh, const bf16_t* __restrict__ wt,
                const float* __restrict__ bias,
                bf16_t* __restrict__ Qb, bf16_t* __restrict__ Kb,
                bf16_t* __restrict__ Vt) {
  __shared__ __align__(128) bf16_t Btile[2][64 * BROWPAD];
  const int lane = threadIdx.x & 31;
  const int wave = threadIdx.x >> 5;
  const int mb   = blockIdx.x & 63;          // 64 M-blocks of 128 rows
  const int ng   = blockIdx.x >> 6;          // 48 N-groups of 64 cols
  const int m0   = mb * 128 + wave * 16;
  const int col0 = ng * 64;
  const int row  = lane & 15;
  const int g    = lane >> 4;
  const int NS   = EE / KTILE;               // 16 K-slabs

  const bf16_t* bsrc = wt + (size_t)col0 * EE;   // 64 rows, stride EE

#if USE_TDM
  if (wave == 0) {
    tdm_load_tile((unsigned int)(uintptr_t)&Btile[0][0], bsrc, EE);
    __builtin_amdgcn_s_wait_tensorcnt(0);
  }
#else
  stage_tile_sync(&Btile[0][0], bsrc, EE, (int)threadIdx.x);
#endif
  __syncthreads();

  v8f acc[4] = {};
  const bf16_t* arow = xh + (size_t)(m0 + row) * EE + 16 * g;

  for (int s = 0; s < NS; ++s) {
    const int cur = s & 1;
    if (s + 1 < NS) {
#if USE_TDM
      if (wave == 0)
        tdm_load_tile((unsigned int)(uintptr_t)&Btile[cur ^ 1][0],
                      bsrc + (s + 1) * KTILE, EE);
#else
      stage_tile_sync(&Btile[cur ^ 1][0], bsrc + (s + 1) * KTILE, EE,
                      (int)threadIdx.x);
#endif
    }
#pragma unroll
    for (int kk2 = 0; kk2 < KTILE; kk2 += 32) {
      v16bf a = load_frag_a(arow + s * KTILE + kk2);
#pragma unroll
      for (int j = 0; j < 4; ++j) {
        const bf16_t* bp = &Btile[cur][(j * 16 + row) * BROWPAD + kk2 + 16 * g];
        acc[j] = wmma_bf16(a, load_frag_b16(bp), acc[j]);
      }
    }
#if USE_TDM
    if (wave == 0 && s + 1 < NS) __builtin_amdgcn_s_wait_tensorcnt(0);
#endif
    __syncthreads();
  }

#pragma unroll
  for (int j = 0; j < 4; ++j) {
#pragma unroll
    for (int r = 0; r < 8; ++r) {
      const int col  = col0 + j * 16 + row;
      const int grow = m0 + r + 8 * g;
      const float v  = acc[j][r] + bias[col];
      const int which  = col >> 10;        // 0=q 1=k 2=v
      const int within = col & 1023;
      const int head   = within >> 6;
      const int hh     = within & 63;
      const int bidx   = grow >> 11;       // /S
      const int sidx   = grow & (SS - 1);
      const size_t bh  = (size_t)(bidx * NH + head);
      const bf16_t hv  = (bf16_t)v;
      if (which == 0)      Qb[(bh * SS + sidx) * HH + hh] = hv;
      else if (which == 1) Kb[(bh * SS + sidx) * HH + hh] = hv;
      else                 Vt[(bh * HH + hh) * SS + sidx] = hv;
    }
  }
}

// ------------------------- stage 2: flash attention -------------------------
// One wave per (b, head, 16-row query tile). Online softmax, O in fp32.

__global__ __launch_bounds__(128)
void k_flash_attn(const bf16_t* __restrict__ Qb, const bf16_t* __restrict__ Kb,
                  const bf16_t* __restrict__ Vt, bf16_t* __restrict__ ctx) {
  __shared__ bf16_t Plds[4][16][32];     // per-wave 16x32 P staging tile
  const int lane = threadIdx.x & 31;
  const int wave = threadIdx.x >> 5;
  const int task = blockIdx.x * 4 + wave;          // 4*16*128 = 8192 tasks
  const int qt   = task & 127;
  const int head = (task >> 7) & (NH - 1);
  const int bidx = task >> 11;
  const int m0   = qt * 16;
  const int row  = lane & 15;
  const int g    = lane >> 4;

  const size_t bh = (size_t)(bidx * NH + head);
  const bf16_t* Qbase = Qb + bh * SS * HH;
  const bf16_t* Kbase = Kb + bh * SS * HH;
  const bf16_t* Vbase = Vt + bh * HH * SS;

  // Q A-fragments for the two 32-wide H chunks (reused across all key blocks)
  const v16bf aq0 = load_frag_a(Qbase + (size_t)(m0 + row) * HH + 0  + 16 * g);
  const v16bf aq1 = load_frag_a(Qbase + (size_t)(m0 + row) * HH + 32 + 16 * g);

  v8f O[4] = {};
  float m_i[8], l_i[8];
#pragma unroll
  for (int r = 0; r < 8; ++r) { m_i[r] = -1e30f; l_i[r] = 0.0f; }

  const float inv_sqrt_h = 0.125f;       // 1/sqrt(64)
  const int nkb = (m0 + 15) / 32 + 1;    // causal: key blocks of 32

  for (int kb = 0; kb < nkb; ++kb) {
    const int k0 = kb * 32;

    // prefetch next key block (global_prefetch_b8)
    if (kb + 1 < nkb) {
      const bf16_t* nk = Kbase + (size_t)(k0 + 32 + row) * HH;
      __builtin_prefetch(nk, 0, 1);
      __builtin_prefetch(nk + (size_t)16 * HH, 0, 1);
    }

    // S = Q K^T : two 16x16 score tiles (keys k0..k0+15 and k0+16..k0+31)
    v8f s0 = {}, s1 = {};
    s0 = wmma_bf16(aq0, load_frag_b(Kbase + (size_t)(k0 + row) * HH + 0  + 16 * g), s0);
    s0 = wmma_bf16(aq1, load_frag_b(Kbase + (size_t)(k0 + row) * HH + 32 + 16 * g), s0);
    s1 = wmma_bf16(aq0, load_frag_b(Kbase + (size_t)(k0 + 16 + row) * HH + 0  + 16 * g), s1);
    s1 = wmma_bf16(aq1, load_frag_b(Kbase + (size_t)(k0 + 16 + row) * HH + 32 + 16 * g), s1);

    // online softmax (per-row; a row lives across one 16-lane group)
    const int c0 = k0 + row, c1 = k0 + 16 + row;
#pragma unroll
    for (int r = 0; r < 8; ++r) {
      const int grow = m0 + r + 8 * g;
      float v0 = (c0 <= grow) ? s0[r] * inv_sqrt_h : -1e30f;
      float v1 = (c1 <= grow) ? s1[r] * inv_sqrt_h : -1e30f;
      float mx = fmaxf(v0, v1);
      mx = fmaxf(mx, __shfl_xor(mx, 1));
      mx = fmaxf(mx, __shfl_xor(mx, 2));
      mx = fmaxf(mx, __shfl_xor(mx, 4));
      mx = fmaxf(mx, __shfl_xor(mx, 8));
      const float mnew = fmaxf(m_i[r], mx);
      const float corr = __expf(m_i[r] - mnew);
      const float p0 = __expf(v0 - mnew);
      const float p1 = __expf(v1 - mnew);
      float rs = p0 + p1;
      rs += __shfl_xor(rs, 1);
      rs += __shfl_xor(rs, 2);
      rs += __shfl_xor(rs, 4);
      rs += __shfl_xor(rs, 8);
      l_i[r] = l_i[r] * corr + rs;
      m_i[r] = mnew;
#pragma unroll
      for (int j = 0; j < 4; ++j) O[j][r] *= corr;
      s0[r] = p0;
      s1[r] = p1;
    }

    // C-layout -> A-layout for P via per-wave LDS tile
#pragma unroll
    for (int r = 0; r < 8; ++r) {
      Plds[wave][r + 8 * g][row]      = (bf16_t)s0[r];
      Plds[wave][r + 8 * g][16 + row] = (bf16_t)s1[r];
    }
    asm volatile("s_wait_dscnt 0" ::: "memory");
    __builtin_amdgcn_wave_barrier();
    const v16bf pa = load_frag_a(&Plds[wave][row][16 * g]);

    // O += P V : V is [H,S] so B-fragments are contiguous
#pragma unroll
    for (int j = 0; j < 4; ++j) {
      const bf16_t* vp = Vbase + (size_t)(j * 16 + row) * SS + k0 + 16 * g;
      O[j] = wmma_bf16(pa, load_frag_b(vp), O[j]);
    }
    __builtin_amdgcn_wave_barrier();
  }

  // normalize and write context [B,S,N*H] as bf16
#pragma unroll
  for (int j = 0; j < 4; ++j) {
#pragma unroll
    for (int r = 0; r < 8; ++r) {
      const int srow = m0 + r + 8 * g;
      const int col  = head * HH + j * 16 + row;
      ctx[((size_t)bidx * SS + srow) * (NH * HH) + col] = (bf16_t)(O[j][r] / l_i[r]);
    }
  }
}

// ------------------------- stage 3: output projection -----------------------
// out = ctx @ W_out + b_out, fp32 result. Same TDM-staged macro-tile scheme.

__global__ __launch_bounds__(256)
void k_out_gemm(const bf16_t* __restrict__ ctx, const bf16_t* __restrict__ wt,
                const float* __restrict__ bias, float* __restrict__ out) {
  __shared__ __align__(128) bf16_t Btile[2][64 * BROWPAD];
  const int lane = threadIdx.x & 31;
  const int wave = threadIdx.x >> 5;
  const int mb   = blockIdx.x & 63;          // 64 M-blocks of 128 rows
  const int ng   = blockIdx.x >> 6;          // 16 N-groups of 64 cols
  const int m0   = mb * 128 + wave * 16;
  const int col0 = ng * 64;
  const int row  = lane & 15;
  const int g    = lane >> 4;
  const int NS   = EE / KTILE;

  const bf16_t* bsrc = wt + (size_t)col0 * EE;

#if USE_TDM
  if (wave == 0) {
    tdm_load_tile((unsigned int)(uintptr_t)&Btile[0][0], bsrc, EE);
    __builtin_amdgcn_s_wait_tensorcnt(0);
  }
#else
  stage_tile_sync(&Btile[0][0], bsrc, EE, (int)threadIdx.x);
#endif
  __syncthreads();

  v8f acc[4] = {};
  const bf16_t* arow = ctx + (size_t)(m0 + row) * EE + 16 * g;

  for (int s = 0; s < NS; ++s) {
    const int cur = s & 1;
    if (s + 1 < NS) {
#if USE_TDM
      if (wave == 0)
        tdm_load_tile((unsigned int)(uintptr_t)&Btile[cur ^ 1][0],
                      bsrc + (s + 1) * KTILE, EE);
#else
      stage_tile_sync(&Btile[cur ^ 1][0], bsrc + (s + 1) * KTILE, EE,
                      (int)threadIdx.x);
#endif
    }
#pragma unroll
    for (int kk2 = 0; kk2 < KTILE; kk2 += 32) {
      v16bf a = load_frag_a(arow + s * KTILE + kk2);
#pragma unroll
      for (int j = 0; j < 4; ++j) {
        const bf16_t* bp = &Btile[cur][(j * 16 + row) * BROWPAD + kk2 + 16 * g];
        acc[j] = wmma_bf16(a, load_frag_b16(bp), acc[j]);
      }
    }
#if USE_TDM
    if (wave == 0 && s + 1 < NS) __builtin_amdgcn_s_wait_tensorcnt(0);
#endif
    __syncthreads();
  }

#pragma unroll
  for (int j = 0; j < 4; ++j) {
#pragma unroll
    for (int r = 0; r < 8; ++r) {
      const int col  = col0 + j * 16 + row;
      const int grow = m0 + r + 8 * g;
      out[(size_t)grow * EE + col] = acc[j][r] + bias[col];
    }
  }
}

// ------------------------------ launcher ------------------------------------

extern "C" void kernel_launch(void* const* d_in, const int* in_sizes, int n_in,
                              void* d_out, int out_size, void* d_ws, size_t ws_size,
                              hipStream_t stream) {
  (void)in_sizes; (void)n_in; (void)out_size; (void)ws_size;
  const float* x     = (const float*)d_in[0];
  const float* W_qkv = (const float*)d_in[1];
  const float* b_qkv = (const float*)d_in[2];
  const float* W_out = (const float*)d_in[3];
  const float* b_out = (const float*)d_in[4];
  float* out = (float*)d_out;

  // workspace carve-up (256B aligned)
  char* ws = (char*)d_ws;
  size_t off = 0;
  auto carve = [&](size_t bytes) -> char* {
    char* p = ws + off;
    off = (off + bytes + 255) & ~(size_t)255;
    return p;
  };
  const size_t n_x   = (size_t)MROWS * EE;           // 8M
  bf16_t* xh     = (bf16_t*)carve(n_x * 2);          // bf16 x
  bf16_t* wqkv_t = (bf16_t*)carve((size_t)QKV3 * EE * 2);
  bf16_t* wout_t = (bf16_t*)carve((size_t)EE * EE * 2);
  bf16_t* Qb     = (bf16_t*)carve(n_x * 2);          // [B,N,S,H]
  bf16_t* Kb     = (bf16_t*)carve(n_x * 2);          // [B,N,S,H]
  bf16_t* Vt     = (bf16_t*)carve(n_x * 2);          // [B,N,H,S]
  bf16_t* ctx    = (bf16_t*)carve(n_x * 2);          // [B,S,N*H]

  // stage 0: converts / weight transposes
  {
    int n = (int)n_x;
    k_cvt_x<<<(n + 255) / 256, 256, 0, stream>>>(x, xh, n);
    int nw = EE * QKV3;
    k_trans_wqkv<<<(nw + 255) / 256, 256, 0, stream>>>(W_qkv, wqkv_t);
    int no = EE * EE;
    k_trans_wout<<<(no + 255) / 256, 256, 0, stream>>>(W_out, wout_t);
  }

  // stage 1: QKV projection (64 M-blocks * 48 N-groups)
  k_qkv_gemm<<<64 * (QKV3 / 64), 256, 0, stream>>>(xh, wqkv_t, b_qkv, Qb, Kb, Vt);

  // stage 2: flash attention (B*N*(S/16) waves, 4 waves/block)
  k_flash_attn<<<BB * NH * (SS / 16) / 4, 128, 0, stream>>>(Qb, Kb, Vt, ctx);

  // stage 3: output projection (64 M-blocks * 16 N-groups)
  k_out_gemm<<<64 * (EE / 64), 256, 0, stream>>>(ctx, wout_t, b_out, out);
}
